// RNNLanguageModel_39573828665776
// MI455X (gfx1250) — compile-verified, matrix operand order checked by reading, not run on hardware
//
#include <hip/hip_runtime.h>
#include <hip/hip_bf16.h>

// ---------------------------------------------------------------------------
// RNN language model forward:
//   emb -> x_proj GEMM -> 256-step tanh recurrence -> logits GEMM
// f16 WMMA (v_wmma_f32_16x16x32_f16) with f32 accumulation throughout.
// Logits GEMM uses double-buffered GLOBAL_LOAD_ASYNC_TO_LDS_B128 staging.
// ---------------------------------------------------------------------------

typedef _Float16 v16h __attribute__((ext_vector_type(16)));
typedef _Float16 v8h  __attribute__((ext_vector_type(8)));
typedef _Float16 v4h  __attribute__((ext_vector_type(4)));
typedef float    v8f  __attribute__((ext_vector_type(8)));

#define V_  32000
#define PAD 1
#define E_  512
#define H_  1024
#define T_  256
#define B_  16
#define TB_ 4096   // T_*B_

// ---------------------------------------------------------------------------
// Fragment loader for 16-bit WMMA operands.
// Works for both A (16x32, vec = row M) and B (32x16, vec = column N):
// per CDNA5 ISA 7.12.2, lanes 0-15 hold K {k0..k0+7, k0+16..k0+23} of
// vector (lane&15); lanes 16-31 hold K {k0+8..k0+15, k0+24..k0+31}.
// `base` points at vector 0; vectors are contiguous runs of `ld` halfs.
// ---------------------------------------------------------------------------
__device__ __forceinline__ v16h load_frag_f16(const _Float16* __restrict__ base,
                                              int ld, int k0) {
  const int lane = threadIdx.x & 31;
  const int v    = lane & 15;
  const int kb   = k0 + ((lane & 16) ? 8 : 0);
  const _Float16* p = base + (size_t)v * ld + kb;
  v8h lo = *(const v8h*)(p);
  v8h hi = *(const v8h*)(p + 16);
  v16h r;
#pragma unroll
  for (int i = 0; i < 8; ++i) { r[i] = lo[i]; r[i + 8] = hi[i]; }
  return r;
}

// ---------------------------------------------------------------------------
// Async global -> LDS copy of 16 bytes per lane (CDNA5 TDM-class async path,
// tracked by ASYNCcnt). lds_off is the wave-group-relative LDS byte address
// (low 32 bits of the flat address per the aperture mapping).
// ---------------------------------------------------------------------------
__device__ __forceinline__ void async_copy_b128(unsigned lds_off,
                                                const void* gaddr) {
  asm volatile("global_load_async_to_lds_b128 %0, %1, off"
               :: "v"(lds_off), "v"(gaddr)
               : "memory");
}

__device__ __forceinline__ unsigned lds_offset_of(const void* p) {
  return (unsigned)(uintptr_t)p;   // addr[31:0] == LDS offset for AS(3) data
}

// ---------------------------------------------------------------------------
// f32 -> f16 weight conversion (vectorized x4)
// ---------------------------------------------------------------------------
__global__ __launch_bounds__(256)
void f32_to_f16_kernel(const float* __restrict__ src, _Float16* __restrict__ dst,
                       int n) {
  int i = (blockIdx.x * 256 + threadIdx.x) * 4;
  if (i + 3 < n) {
    float4 v = *(const float4*)(src + i);
    v4h o = { (_Float16)v.x, (_Float16)v.y, (_Float16)v.z, (_Float16)v.w };
    *(v4h*)(dst + i) = o;
  } else {
    for (; i < n; ++i) dst[i] = (_Float16)src[i];
  }
}

// ---------------------------------------------------------------------------
// Embedding gather + x_proj = emb @ W_ih^T + b_ih   ([4096,512]x[512,1024])
// One wave per 16x16 output tile; A gathered from emb_table (PAD masked).
// ---------------------------------------------------------------------------
__global__ __launch_bounds__(32)
void embed_xproj_kernel(const int* __restrict__ text,
                        const float* __restrict__ emb_table,
                        const _Float16* __restrict__ Wih,   // [H,E] f16 row-major
                        const float* __restrict__ b_ih,
                        float* __restrict__ x_proj)         // [TB,H] f32
{
  const int lane = threadIdx.x;
  const int n0 = blockIdx.x * 16;
  const int m0 = blockIdx.y * 16;
  const int r   = lane & 15;
  const int hi8 = (lane & 16) ? 8 : 0;

  const int tok = text[m0 + r];                 // token for this lane's A-row
  const float s = (tok == PAD) ? 0.0f : 1.0f;   // padding_idx mask
  const float* arow = emb_table + (size_t)tok * E_;

  v8f acc = {};
  for (int k0 = 0; k0 < E_; k0 += 32) {
    const int kb = k0 + hi8;
    float4 x0 = *(const float4*)(arow + kb);
    float4 x1 = *(const float4*)(arow + kb + 4);
    float4 x2 = *(const float4*)(arow + kb + 16);
    float4 x3 = *(const float4*)(arow + kb + 20);
    v16h a;
    a[0] = (_Float16)(x0.x * s);  a[1] = (_Float16)(x0.y * s);
    a[2] = (_Float16)(x0.z * s);  a[3] = (_Float16)(x0.w * s);
    a[4] = (_Float16)(x1.x * s);  a[5] = (_Float16)(x1.y * s);
    a[6] = (_Float16)(x1.z * s);  a[7] = (_Float16)(x1.w * s);
    a[8] = (_Float16)(x2.x * s);  a[9] = (_Float16)(x2.y * s);
    a[10] = (_Float16)(x2.z * s); a[11] = (_Float16)(x2.w * s);
    a[12] = (_Float16)(x3.x * s); a[13] = (_Float16)(x3.y * s);
    a[14] = (_Float16)(x3.z * s); a[15] = (_Float16)(x3.w * s);

    v16h b = load_frag_f16(Wih + (size_t)n0 * E_, E_, k0);  // B[k][n]=Wih[n][k]
    acc = __builtin_amdgcn_wmma_f32_16x16x32_f16(false, a, false, b,
                                                 (short)0, acc, false, false);
  }
  const int col = n0 + r;
  const float bias = b_ih[col];
#pragma unroll
  for (int i = 0; i < 8; ++i) {
    const int m = m0 + i + hi8;
    x_proj[(size_t)m * H_ + col] = acc[i] + bias;
  }
}

// ---------------------------------------------------------------------------
// Device-wide barrier (agent scope) for the persistent recurrence kernel.
// bar[0] = arrival counter, bar[1] = generation.
// ---------------------------------------------------------------------------
__device__ __forceinline__ void grid_barrier(unsigned* bar, unsigned nb) {
  __builtin_amdgcn_fence(__ATOMIC_RELEASE, "agent");
  __syncthreads();
  if (threadIdx.x == 0) {
    unsigned g = __hip_atomic_load(bar + 1, __ATOMIC_RELAXED,
                                   __HIP_MEMORY_SCOPE_AGENT);
    if (__hip_atomic_fetch_add(bar, 1u, __ATOMIC_ACQ_REL,
                               __HIP_MEMORY_SCOPE_AGENT) == nb - 1) {
      __hip_atomic_store(bar, 0u, __ATOMIC_RELAXED, __HIP_MEMORY_SCOPE_AGENT);
      __hip_atomic_fetch_add(bar + 1, 1u, __ATOMIC_RELEASE,
                             __HIP_MEMORY_SCOPE_AGENT);
    } else {
      while (__hip_atomic_load(bar + 1, __ATOMIC_RELAXED,
                               __HIP_MEMORY_SCOPE_AGENT) == g)
        __builtin_amdgcn_s_sleep(2);
    }
  }
  __syncthreads();
  __builtin_amdgcn_fence(__ATOMIC_ACQUIRE, "agent");
}

__global__ void init_barrier_kernel(unsigned* bar) {
  if (threadIdx.x < 2) bar[threadIdx.x] = 0u;
}

// ---------------------------------------------------------------------------
// Persistent recurrence: 64 single-wave blocks, each owns a 16-wide column
// tile of H. h_t = tanh(x_proj_t + h_{t-1} @ W_hh^T + b_hh). Hidden state
// kept in f16 (it only feeds f16 WMMA consumers). Grid barrier per step.
// ---------------------------------------------------------------------------
__global__ __launch_bounds__(32)
void rnn_step_kernel(const float* __restrict__ x_proj,   // [T,B,H] f32
                     const _Float16* __restrict__ Whh,   // [H,H] f16 row-major
                     const float* __restrict__ b_hh,
                     _Float16* __restrict__ hout,        // [T,B,H] f16
                     unsigned* __restrict__ bar)
{
  const int lane = threadIdx.x;
  const int n0  = blockIdx.x * 16;
  const int r   = lane & 15;
  const int hi8 = (lane & 16) ? 8 : 0;
  const int col = n0 + r;
  const float bias = b_hh[col];
  const _Float16* wb = Whh + (size_t)n0 * H_;

  for (int t = 0; t < T_; ++t) {
    v8f acc = {};
    if (t > 0) {
      const _Float16* hprev = hout + (size_t)(t - 1) * (B_ * H_);
#pragma unroll 4
      for (int k0 = 0; k0 < H_; k0 += 32) {
        v16h a = load_frag_f16(hprev, H_, k0);   // A row = batch index
        v16h b = load_frag_f16(wb, H_, k0);      // B[k][n] = Whh[n][k]
        acc = __builtin_amdgcn_wmma_f32_16x16x32_f16(false, a, false, b,
                                                     (short)0, acc,
                                                     false, false);
      }
    }
    const float* xp = x_proj + (size_t)t * (B_ * H_);
    _Float16* ho    = hout   + (size_t)t * (B_ * H_);
#pragma unroll
    for (int i = 0; i < 8; ++i) {
      const int bb = i + hi8;                     // batch row of C tile
      float v = acc[i] + xp[(size_t)bb * H_ + col] + bias;
      ho[(size_t)bb * H_ + col] = (_Float16)tanhf(v);
    }
    grid_barrier(bar, gridDim.x);
  }
}

// ---------------------------------------------------------------------------
// Logits GEMM: [4096,1024] x [1024, 32000] (+b_out). 8 waves per block,
// 128x128 output tile, K staged through double-buffered LDS via
// GLOBAL_LOAD_ASYNC_TO_LDS_B128 (ASYNCcnt), overlapping DMA with WMMA.
// ---------------------------------------------------------------------------
__global__ __launch_bounds__(256)
void logits_kernel(const _Float16* __restrict__ Hf,    // [TB,H] f16
                   const _Float16* __restrict__ Wout,  // [V,H]  f16 row-major
                   const float* __restrict__ b_out,
                   float* __restrict__ out)            // [TB,V] f32
{
  __shared__ _Float16 sA[2][128 * 32];   // rows m, 32 halfs each
  __shared__ _Float16 sB[2][128 * 32];   // cols n (rows of Wout), 32 halfs

  const int tid  = threadIdx.x;
  const int lane = tid & 31;
  const int wave = tid >> 5;
  const int n0 = blockIdx.x * 128;
  const int m0 = blockIdx.y * 128;
  const int r   = lane & 15;
  const int hi8 = (lane & 16) ? 8 : 0;

  const int ldRow  = tid >> 1;         // 0..127
  const int ldHalf = (tid & 1) * 16;   // 0 or 16 (halfs)

  v8f acc[8];
#pragma unroll
  for (int s = 0; s < 8; ++s) acc[s] = {};

  const _Float16* aSrc = Hf   + (size_t)(m0 + ldRow) * H_ + ldHalf;
  const _Float16* bSrc = Wout + (size_t)(n0 + ldRow) * H_ + ldHalf;

  const unsigned aOff0 = lds_offset_of(&sA[0][ldRow * 32 + ldHalf]);
  const unsigned aOff1 = lds_offset_of(&sA[1][ldRow * 32 + ldHalf]);
  const unsigned bOff0 = lds_offset_of(&sB[0][ldRow * 32 + ldHalf]);
  const unsigned bOff1 = lds_offset_of(&sB[1][ldRow * 32 + ldHalf]);

  // issue 4 async b128 ops (32B of A + 32B of B) for K-chunk k0 into buffer b
  auto issue = [&](int b, int k0) {
    const unsigned ao = b ? aOff1 : aOff0;
    const unsigned bo = b ? bOff1 : bOff0;
    async_copy_b128(ao,      aSrc + k0);
    async_copy_b128(ao + 16, aSrc + k0 + 8);
    async_copy_b128(bo,      bSrc + k0);
    async_copy_b128(bo + 16, bSrc + k0 + 8);
  };

  constexpr int NK = H_ / 32;          // 32 K-chunks
  issue(0, 0);

  for (int i = 0; i < NK; ++i) {
    const int cur = i & 1;
    if (i + 1 < NK) {
      issue(cur ^ 1, (i + 1) * 32);
      // in-order ASYNCcnt: wait until only the 4 just-issued remain
      asm volatile("s_wait_asynccnt 0x4" ::: "memory");
    } else {
      asm volatile("s_wait_asynccnt 0x0" ::: "memory");
    }
    __syncthreads();

    v16h a = load_frag_f16(&sA[cur][(wave * 16) * 32], 32, 0);
#pragma unroll
    for (int s = 0; s < 8; ++s) {
      v16h b = load_frag_f16(&sB[cur][(s * 16) * 32], 32, 0);
      acc[s] = __builtin_amdgcn_wmma_f32_16x16x32_f16(false, a, false, b,
                                                      (short)0, acc[s],
                                                      false, false);
    }
    __syncthreads();   // all waves done reading buf `cur` before re-issue
  }

  const int mbase = m0 + wave * 16 + hi8;
#pragma unroll
  for (int s = 0; s < 8; ++s) {
    const int col = n0 + s * 16 + r;
    const float bias = b_out[col];
#pragma unroll
    for (int i = 0; i < 8; ++i) {
      out[(size_t)(mbase + i) * V_ + col] = acc[s][i] + bias;
    }
  }
}

// ---------------------------------------------------------------------------
// Host-side launcher
// ---------------------------------------------------------------------------
extern "C" void kernel_launch(void* const* d_in, const int* in_sizes, int n_in,
                              void* d_out, int out_size, void* d_ws, size_t ws_size,
                              hipStream_t stream) {
  const int*   text      = (const int*)  d_in[0];
  // d_in[1] = hidden_0 (all zeros; recurrence handles t==0 analytically)
  const float* emb_table = (const float*)d_in[2];
  const float* W_ih      = (const float*)d_in[3];
  const float* W_hh      = (const float*)d_in[4];
  const float* b_ih      = (const float*)d_in[5];
  const float* b_hh      = (const float*)d_in[6];
  const float* W_out     = (const float*)d_in[7];
  const float* b_out     = (const float*)d_in[8];
  float* out = (float*)d_out;

  char* ws = (char*)d_ws;
  // workspace layout (256B aligned)
  _Float16* Wih_h  = (_Float16*)(ws + 0);                    //  1 MB
  _Float16* Whh_h  = (_Float16*)(ws + 1048576);              //  2 MB
  _Float16* Wout_h = (_Float16*)(ws + 3145728);              // 65.5 MB
  float*    x_proj = (float*)   (ws + 68681728);             // 16.8 MB
  _Float16* hout   = (_Float16*)(ws + 85458944);             //  8.4 MB
  unsigned* bar    = (unsigned*)(ws + 93847552);

  // 1) weight conversions f32 -> f16
  {
    int n = H_ * E_;        // 524288
    f32_to_f16_kernel<<<(n / 4 + 255) / 256, 256, 0, stream>>>(W_ih, Wih_h, n);
  }
  {
    int n = H_ * H_;        // 1048576
    f32_to_f16_kernel<<<(n / 4 + 255) / 256, 256, 0, stream>>>(W_hh, Whh_h, n);
  }
  {
    int n = V_ * H_;        // 32768000
    f32_to_f16_kernel<<<(n / 4 + 255) / 256, 256, 0, stream>>>(W_out, Wout_h, n);
  }

  // 2) embedding + input projection
  embed_xproj_kernel<<<dim3(H_ / 16, TB_ / 16), 32, 0, stream>>>(
      text, emb_table, Wih_h, b_ih, x_proj);

  // 3) recurrence (persistent grid + device barrier)
  init_barrier_kernel<<<1, 32, 0, stream>>>(bar);
  rnn_step_kernel<<<H_ / 16, 32, 0, stream>>>(x_proj, Whh_h, b_hh, hout, bar);

  // 4) logits GEMM
  logits_kernel<<<dim3(V_ / 128, TB_ / 128), 256, 0, stream>>>(
      hout, Wout_h, b_out, out);
}
